// MultiheadAttention_23922967838736
// MI455X (gfx1250) — matrix operand order repaired and checked by reading, not compiled, not run on hardware
//
#include <hip/hip_runtime.h>
#include <math.h>

// ---------------------------------------------------------------------------
// MHA forward for MI455X (gfx1250, wave32, WMMA bf16 w/ fp32 accumulate)
// B=2, L=2048, D=1024, H=16, DH=64
// ---------------------------------------------------------------------------

typedef __attribute__((ext_vector_type(16))) __bf16 v16bf;
typedef __attribute__((ext_vector_type(8)))  __bf16 v8bf;
typedef __attribute__((ext_vector_type(8)))  float  v8f;
typedef __attribute__((ext_vector_type(4)))  unsigned int v4u;
typedef int v4i_ __attribute__((vector_size(16)));   // matches builtin param type

#define WMMA_BF16(a, b, c) \
  __builtin_amdgcn_wmma_f32_16x16x32_bf16(false, (a), false, (b), (short)0, (c), false, false)

static __device__ __forceinline__ v16bf cat8(v8bf lo, v8bf hi) {
  v16bf r;
#pragma unroll
  for (int i = 0; i < 8; ++i) { r[i] = lo[i]; r[i + 8] = hi[i]; }
  return r;
}

#define BB 2
#define LL 2048
#define DD 1024
#define HH 16
#define DH 64
#define MM (BB * LL)   // 4096

// ---------------------------------------------------------------------------
// Async global->LDS staging (CDNA5 GLOBAL_LOAD_ASYNC_TO_LDS_B128, ASYNCcnt).
// Builtin signature (from hipcc diagnostic): arg0 = v4i addrspace(1)*,
// arg1 = LDS side, then two immediate ints (offset, cpol).
// ---------------------------------------------------------------------------
#if __has_builtin(__builtin_amdgcn_global_load_async_to_lds_b128)
#define HAVE_ASYNC_LDS 1
#else
#define HAVE_ASYNC_LDS 0
#endif

typedef v4i_ __attribute__((address_space(1)))* gv4i_p;
typedef v4i_ __attribute__((address_space(3)))* lv4i_p;

static __device__ __forceinline__ void async_copy16(const __bf16* g, __bf16* l) {
#if HAVE_ASYNC_LDS
  __builtin_amdgcn_global_load_async_to_lds_b128(
      (gv4i_p)(unsigned long long)g,
      (lv4i_p)(unsigned int)(unsigned long long)l, 0, 0);
#else
  *(v4u*)l = *(const v4u*)g;
#endif
}

static __device__ __forceinline__ void async_wait0() {
#if HAVE_ASYNC_LDS
#if __has_builtin(__builtin_amdgcn_s_wait_asynccnt)
  __builtin_amdgcn_s_wait_asynccnt(0);
#else
  asm volatile("s_wait_asynccnt 0x0" ::: "memory");
#endif
#endif
}

// ---------------------------------------------------------------------------
// fp32 -> bf16 conversion (grid-stride), used for x
// ---------------------------------------------------------------------------
__global__ void f32_to_bf16_k(const float* __restrict__ src, __bf16* __restrict__ dst, int n) {
  int i = blockIdx.x * blockDim.x + threadIdx.x;
  int stride = gridDim.x * blockDim.x;
  for (; i < n; i += stride) dst[i] = (__bf16)src[i];
}

// ---------------------------------------------------------------------------
// Weight convert + transpose: W[k][n] f32 -> Wt[n][k] bf16 (DD x DD).
// Coalesced reads; scattered 2B writes (only 2MB per weight, L2-resident).
// ---------------------------------------------------------------------------
__global__ void w_transpose_bf16_k(const float* __restrict__ src, __bf16* __restrict__ dst) {
  int idx = blockIdx.x * blockDim.x + threadIdx.x;   // 0 .. DD*DD-1
  int k = idx >> 10;
  int n = idx & (DD - 1);
  dst[(size_t)n * DD + k] = (__bf16)src[idx];
}

// ---------------------------------------------------------------------------
// GEMM: C[M=4096, N=1024] = A_bf16 @ Wt_bf16^T + bias, scatter per MODE.
//   MODE 0: dst_bf[B,H,L,DH]  (Q with scale=0.125, K with scale=1)
//   MODE 1: dst_bf[B,H,DH,L]  (V transposed for attention B-fragments)
//   MODE 2: dst_f [M,N] fp32  (final output projection)
// Block: 256 threads (8 waves). Block tile 128(M) x 128(N), K-step 32.
// Wave grid 4x2; each wave computes a 32x64 C tile (2x4 WMMA accumulators):
// 8 WMMAs per K-step vs 6 fragment loads. LDS strides padded to 48 elems
// (96B) for 16B-aligned b128 fragment reads with reduced bank conflicts.
// Both A and B tiles are staged with async global->LDS b128 copies.
// ---------------------------------------------------------------------------
template <int MODE>
__global__ __launch_bounds__(256) void gemm_bf16_k(
    const __bf16* __restrict__ A, const __bf16* __restrict__ Wt,
    const float* __restrict__ bias, __bf16* __restrict__ dst_bf,
    float* __restrict__ dst_f, float scale) {
  __shared__ __bf16 As[128 * 48];  // [m][k], k contiguous
  __shared__ __bf16 Bs[128 * 48];  // [n][k], k contiguous (Wt is pre-transposed)

  const int tid  = threadIdx.x;
  const int wave = tid >> 5;
  const int lane = tid & 31;
  const int hi   = (lane >> 4) & 1;
  const int l15  = lane & 15;
  const int mBase = blockIdx.y * 128;
  const int nBase = blockIdx.x * 128;
  const int wm = wave >> 1;   // 0..3 -> m offset wm*32
  const int wn = wave & 1;    // 0..1 -> n offset wn*64

  v8f acc[2][4];
#pragma unroll
  for (int i = 0; i < 2; ++i)
#pragma unroll
    for (int j = 0; j < 4; ++j)
#pragma unroll
      for (int r = 0; r < 8; ++r) acc[i][j][r] = 0.0f;

  const int klo  = hi ? 8 : 0;
  const int koff = hi ? 16 : 0;

  for (int kb = 0; kb < DD; kb += 32) {
    // Stage A (128x32) and B (128x32) tiles: 2 b128 chunks each per thread.
#pragma unroll
    for (int i = 0; i < 2; ++i) {
      int c = tid + i * 256;            // 0..511
      int row = c >> 2;                 // 0..127
      int kc = (c & 3) * 8;             // 0,8,16,24
      async_copy16(A  + (size_t)(mBase + row) * DD + kb + kc, &As[row * 48 + kc]);
      async_copy16(Wt + (size_t)(nBase + row) * DD + kb + kc, &Bs[row * 48 + kc]);
    }
    async_wait0();
    __syncthreads();

    // A fragments (16x32): lane=row; elems 0-7: K=klo..+7, 8-15: K=klo+16..+7
    v16bf af[2], bfr[4];
#pragma unroll
    for (int i = 0; i < 2; ++i) {
      int row = wm * 32 + i * 16 + l15;
      v8bf lo = *(const v8bf*)(&As[row * 48 + klo]);
      v8bf hh = *(const v8bf*)(&As[row * 48 + klo + 16]);
      af[i] = cat8(lo, hh);
    }
    // B fragments (32x16): lane=col; elems 0-7: K=klo..+7, 8-15: K=klo+16..+7
#pragma unroll
    for (int j = 0; j < 4; ++j) {
      int n = wn * 64 + j * 16 + l15;
      v8bf lo = *(const v8bf*)(&Bs[n * 48 + klo]);
      v8bf hh = *(const v8bf*)(&Bs[n * 48 + klo + 16]);
      bfr[j] = cat8(lo, hh);
    }
#pragma unroll
    for (int i = 0; i < 2; ++i)
#pragma unroll
      for (int j = 0; j < 4; ++j)
        acc[i][j] = WMMA_BF16(af[i], bfr[j], acc[i][j]);
    __syncthreads();
  }

  // Epilogue (branch-free: MODE is a template parameter)
#pragma unroll
  for (int i = 0; i < 2; ++i) {
#pragma unroll
    for (int j = 0; j < 4; ++j) {
      int gcol = nBase + wn * 64 + j * 16 + l15;
      float bvv = bias[gcol];
      int h  = gcol >> 6;
      int dh = gcol & 63;
#pragma unroll
      for (int r = 0; r < 8; ++r) {
        int grow = mBase + wm * 32 + i * 16 + r + (hi ? 8 : 0);
        float v = (acc[i][j][r] + bvv) * scale;
        int b  = grow >> 11;        // / 2048
        int ql = grow & 2047;
        if (MODE == 0) {
          dst_bf[((size_t)(b * HH + h) * LL + ql) * DH + dh] = (__bf16)v;
        } else if (MODE == 1) {
          dst_bf[((size_t)(b * HH + h) * DH + dh) * LL + ql] = (__bf16)v;
        } else {
          dst_f[(size_t)grow * DD + gcol] = v;
        }
      }
    }
  }
}

// ---------------------------------------------------------------------------
// Flash attention: grid (L/128, B*H). 256 threads = 8 waves; wave w owns 16
// queries. Online softmax; QK^T and PV via WMMA; P goes C-layout -> A-layout
// through a per-wave LDS tile. Q pre-scaled by 1/sqrt(DH) in projection.
// Q,K: [B,H,L,DH] bf16.  Vt: [B,H,DH,L] bf16.  Out: [B*L, D] bf16.
// ---------------------------------------------------------------------------
__global__ __launch_bounds__(256) void attn_k(
    const __bf16* __restrict__ Q, const __bf16* __restrict__ K,
    const __bf16* __restrict__ Vt, __bf16* __restrict__ Out) {
  __shared__ __bf16 Plds[8 * 16 * 48];

  const int tid  = threadIdx.x;
  const int wave = tid >> 5;
  const int lane = tid & 31;
  const int hi   = (lane >> 4) & 1;
  const int l15  = lane & 15;
  const int bh   = blockIdx.y;                   // 0..31
  const int qBase = blockIdx.x * 128 + wave * 16;

  const __bf16* Qb = Q  + (size_t)bh * LL * DH;
  const __bf16* Kb = K  + (size_t)bh * LL * DH;
  const __bf16* Vb = Vt + (size_t)bh * DH * LL;
  __bf16* Pw = &Plds[wave * 16 * 48];

  const int klo  = hi ? 8 : 0;
  const int koff = hi ? 16 : 0;

  // Q fragments over DH=64: 2 chunks of K=32
  v16bf qf[2];
  {
    int q = qBase + l15;
#pragma unroll
    for (int c = 0; c < 2; ++c) {
      v8bf lo = *(const v8bf*)(Qb + (size_t)q * DH + c * 32 + klo);
      v8bf hh = *(const v8bf*)(Qb + (size_t)q * DH + c * 32 + klo + 16);
      qf[c] = cat8(lo, hh);
    }
  }

  v8f o[4];
#pragma unroll
  for (int t = 0; t < 4; ++t)
#pragma unroll
    for (int r = 0; r < 8; ++r) o[t][r] = 0.0f;
  float mrow[8], lrow[8];
#pragma unroll
  for (int r = 0; r < 8; ++r) { mrow[r] = -INFINITY; lrow[r] = 0.0f; }

  for (int kb = 0; kb < LL; kb += 32) {
    // Prefetch next key/value block into cache (global_prefetch_b8)
    if (kb + 32 < LL) {
      __builtin_prefetch(Kb + (size_t)(kb + 32 + lane) * DH, 0, 3);
      __builtin_prefetch(Vb + (size_t)(lane << 1) * LL + kb + 32, 0, 3);
    }

    // --- scores S[16x32] = Q @ K^T : two 16x16 tiles, 2 dh-chunks each ---
    v8f s0, s1;
#pragma unroll
    for (int r = 0; r < 8; ++r) { s0[r] = 0.0f; s1[r] = 0.0f; }
#pragma unroll
    for (int c = 0; c < 2; ++c) {
      int key0 = kb + l15;
      int key1 = kb + 16 + l15;
      v8bf lo0 = *(const v8bf*)(Kb + (size_t)key0 * DH + c * 32 + klo);
      v8bf hh0 = *(const v8bf*)(Kb + (size_t)key0 * DH + c * 32 + klo + 16);
      v8bf lo1 = *(const v8bf*)(Kb + (size_t)key1 * DH + c * 32 + klo);
      v8bf hh1 = *(const v8bf*)(Kb + (size_t)key1 * DH + c * 32 + klo + 16);
      s0 = WMMA_BF16(qf[c], cat8(lo0, hh0), s0);
      s1 = WMMA_BF16(qf[c], cat8(lo1, hh1), s1);
    }

    // --- online softmax (per element-row r; half-wave reductions) ---
#pragma unroll
    for (int r = 0; r < 8; ++r) {
      float a0 = s0[r], a1 = s1[r];
      float v = fmaxf(a0, a1);
      v = fmaxf(v, __shfl_xor(v, 1, 32));
      v = fmaxf(v, __shfl_xor(v, 2, 32));
      v = fmaxf(v, __shfl_xor(v, 4, 32));
      v = fmaxf(v, __shfl_xor(v, 8, 32));
      float mnew  = fmaxf(mrow[r], v);
      float alpha = __expf(mrow[r] - mnew);
      float p0 = __expf(a0 - mnew);
      float p1 = __expf(a1 - mnew);
      float rs = p0 + p1;
      rs += __shfl_xor(rs, 1, 32);
      rs += __shfl_xor(rs, 2, 32);
      rs += __shfl_xor(rs, 4, 32);
      rs += __shfl_xor(rs, 8, 32);
      lrow[r] = lrow[r] * alpha + rs;
      mrow[r] = mnew;
#pragma unroll
      for (int t = 0; t < 4; ++t) o[t][r] *= alpha;
      int prow = r + hi * 8;
      Pw[prow * 48 + l15]      = (__bf16)p0;
      Pw[prow * 48 + 16 + l15] = (__bf16)p1;
    }

    // --- P (C layout in LDS) -> A fragment ---
    v8bf plo = *(const v8bf*)(&Pw[l15 * 48 + klo]);
    v8bf phi = *(const v8bf*)(&Pw[l15 * 48 + klo + 16]);
    v16bf pf = cat8(plo, phi);

    // --- O += P @ V : 4 output tiles over dh; V^T gives contiguous K reads ---
#pragma unroll
    for (int t = 0; t < 4; ++t) {
      int dh = t * 16 + l15;
      v8bf lo = *(const v8bf*)(Vb + (size_t)dh * LL + kb + klo);
      v8bf hh = *(const v8bf*)(Vb + (size_t)dh * LL + kb + klo + 16);
      o[t] = WMMA_BF16(pf, cat8(lo, hh), o[t]);
    }
  }

  // --- finalize: O / l, store to [B*L, D] bf16 (merged heads) ---
  const int b = bh >> 4;
  const int h = bh & 15;
#pragma unroll
  for (int r = 0; r < 8; ++r) {
    float inv = 1.0f / lrow[r];
    int grow = (b << 11) + qBase + r + hi * 8;
#pragma unroll
    for (int t = 0; t < 4; ++t) {
      int gcol = h * 64 + t * 16 + l15;
      Out[(size_t)grow * DD + gcol] = (__bf16)(o[t][r] * inv);
    }
  }
}

// ---------------------------------------------------------------------------
// Host launcher
// ---------------------------------------------------------------------------
extern "C" void kernel_launch(void* const* d_in, const int* in_sizes, int n_in,
                              void* d_out, int out_size, void* d_ws, size_t ws_size,
                              hipStream_t stream) {
  (void)in_sizes; (void)n_in; (void)out_size; (void)ws_size;
  const float* x  = (const float*)d_in[0];
  const float* Wq = (const float*)d_in[1];
  const float* bq = (const float*)d_in[2];
  const float* Wk = (const float*)d_in[3];
  const float* bk = (const float*)d_in[4];
  const float* Wv = (const float*)d_in[5];
  const float* bv = (const float*)d_in[6];
  const float* Wo = (const float*)d_in[7];
  const float* bo = (const float*)d_in[8];

  char* ws = (char*)d_ws;
  size_t off = 0;
  __bf16* Xb  = (__bf16*)(ws + off); off += (size_t)MM * DD * 2;  // x bf16
  __bf16* Wqt = (__bf16*)(ws + off); off += (size_t)DD * DD * 2;  // W^T bf16
  __bf16* Wkt = (__bf16*)(ws + off); off += (size_t)DD * DD * 2;
  __bf16* Wvt = (__bf16*)(ws + off); off += (size_t)DD * DD * 2;
  __bf16* Wot = (__bf16*)(ws + off); off += (size_t)DD * DD * 2;
  __bf16* Qb  = (__bf16*)(ws + off); off += (size_t)MM * DD * 2;  // [B,H,L,DH]
  __bf16* Kb  = (__bf16*)(ws + off); off += (size_t)MM * DD * 2;  // [B,H,L,DH]
  __bf16* Vtb = (__bf16*)(ws + off); off += (size_t)MM * DD * 2;  // [B,H,DH,L]
  __bf16* Ob  = (__bf16*)(ws + off); off += (size_t)MM * DD * 2;  // [B*L, D]

  // Stage 1: convert x; convert+transpose weights
  f32_to_bf16_k<<<2048, 256, 0, stream>>>(x, Xb, MM * DD);
  w_transpose_bf16_k<<<DD * DD / 256, 256, 0, stream>>>(Wq, Wqt);
  w_transpose_bf16_k<<<DD * DD / 256, 256, 0, stream>>>(Wk, Wkt);
  w_transpose_bf16_k<<<DD * DD / 256, 256, 0, stream>>>(Wv, Wvt);
  w_transpose_bf16_k<<<DD * DD / 256, 256, 0, stream>>>(Wo, Wot);

  // Stage 2: Q/K/V projections (Q pre-scaled by 1/sqrt(DH))
  dim3 ggrid(DD / 128, MM / 128);   // (8, 32)
  gemm_bf16_k<0><<<ggrid, 256, 0, stream>>>(Xb, Wqt, bq, Qb,  nullptr, 0.125f);
  gemm_bf16_k<0><<<ggrid, 256, 0, stream>>>(Xb, Wkt, bk, Kb,  nullptr, 1.0f);
  gemm_bf16_k<1><<<ggrid, 256, 0, stream>>>(Xb, Wvt, bv, Vtb, nullptr, 1.0f);

  // Stage 3: flash attention
  attn_k<<<dim3(LL / 128, BB * HH), 256, 0, stream>>>(Qb, Kb, Vtb, Ob);

  // Stage 4: output projection -> fp32 d_out
  gemm_bf16_k<2><<<ggrid, 256, 0, stream>>>(Ob, Wot, bo, nullptr, (float*)d_out, 1.0f);
}
